// MoeMLP_13520557048401
// MI455X (gfx1250) — matrix operand order repaired and checked by reading, not compiled, not run on hardware
//
#include <hip/hip_runtime.h>
#include <hip/hip_bf16.h>
#include <math.h>

// ---------------- problem constants (match reference) ----------------
constexpr int Bb  = 8;
constexpr int Ss  = 2048;
constexpr int Dm  = 1024;          // n_embd
constexpr int Ee  = 8;             // experts
constexpr int Kk  = 2;             // top-k
constexpr int BLK = 128;           // token block
constexpr int DFF = 2048;          // per-expert ffn width
constexpr int T   = Bb * Ss;       // 16384 tokens
constexpr int NR  = T * Kk;        // 32768 routed rows
constexpr int MAXB = NR / BLK + Ee;      // 264 max padded blocks
constexpr int CAP  = MAXB * BLK;         // 33792 padded rows

// ---------------- workspace layout (bytes) ----------------
constexpr size_t OFF_COUNTS = 0;                       // 8 * i32
constexpr size_t OFF_FILL   = 64;                      // 8 * i32
constexpr size_t OFF_PSUM   = 128;                     // 8 * f32
constexpr size_t OFF_ZACC   = 192;                     // 1 * f32
constexpr size_t OFF_OFFPAD = 256;                     // 9 * i32
constexpr size_t OFF_META   = 320;                     // n_blocks, cap
constexpr size_t OFF_BLKEXP = 384;                     // MAXB * i32
constexpr size_t OFF_SEL    = 2048;                    // NR * i32
constexpr size_t OFF_RW     = OFF_SEL + (size_t)NR * 4;
constexpr size_t OFF_POS    = OFF_RW  + (size_t)NR * 4;
constexpr size_t OFF_XG     = ((OFF_POS + (size_t)NR * 4 + 255) / 256) * 256; // CAP x Dm bf16
constexpr size_t OFF_W1T    = OFF_XG  + (size_t)CAP * Dm  * 2;  // transpose(w1): [Ee*DFF][Dm] bf16
constexpr size_t OFF_W2T    = OFF_W1T + (size_t)Dm  * Ee * DFF * 2; // transpose(w2): [Dm][Ee*DFF] bf16
constexpr size_t OFF_H      = OFF_W2T + (size_t)Ee  * DFF * Dm * 2; // CAP x DFF bf16
constexpr size_t OFF_Y      = OFF_H   + (size_t)CAP * DFF * 2;      // CAP x Dm bf16

typedef __attribute__((ext_vector_type(16))) __bf16 v16bf;
typedef __attribute__((ext_vector_type(8)))  float  v8f;
typedef __attribute__((ext_vector_type(4)))  int    v4i;

// ---------------- gfx1250 async global->LDS copy (ASYNCcnt path) ----------------
#if defined(__HIP_DEVICE_COMPILE__) && __has_builtin(__builtin_amdgcn_global_load_async_to_lds_b128)
#define ASYNC_LDS 1
#else
#define ASYNC_LDS 0
#endif

__device__ __forceinline__ void cp_b128(void* lds, const void* g) {
#if ASYNC_LDS
    __builtin_amdgcn_global_load_async_to_lds_b128(
        (__attribute__((address_space(1))) v4i*)(g),
        (__attribute__((address_space(3))) v4i*)(lds), 0, 0);
#else
    *reinterpret_cast<v4i*>(lds) = *reinterpret_cast<const v4i*>(g);
#endif
}

__device__ __forceinline__ void wait_async_all() {
#if ASYNC_LDS
#if __has_builtin(__builtin_amdgcn_s_wait_asynccnt)
    __builtin_amdgcn_s_wait_asynccnt(0);
#else
    asm volatile("s_wait_asynccnt 0" ::: "memory");
#endif
#endif
}

// ---------------- small init ----------------
__global__ void moe_init_small(int* counts, int* fill, float* psum, float* zacc) {
    int i = threadIdx.x;
    if (i < Ee) { counts[i] = 0; fill[i] = 0; psum[i] = 0.f; }
    if (i == 0) zacc[0] = 0.f;
}

__global__ void moe_zero_u4(uint4* p, long n) {
    long i = (long)blockIdx.x * blockDim.x + threadIdx.x;
    if (i < n) { uint4 z; z.x = z.y = z.z = z.w = 0u; p[i] = z; }
}

// ---------------- tiled transpose + f32->bf16 convert: out[c*R + r] = in[r*C + c] ----------------
__global__ __launch_bounds__(256)
void moe_transpose_cvt(const float* __restrict__ in, __bf16* __restrict__ out,
                       int R, int C) {
    __shared__ float tile[32][33];
    int c0 = blockIdx.x * 32;
    int r0 = blockIdx.y * 32;
    int lx = threadIdx.x & 31, ly = threadIdx.x >> 5;   // 32 x 8
#pragma unroll
    for (int i = ly; i < 32; i += 8)
        tile[i][lx] = in[(size_t)(r0 + i) * C + c0 + lx];
    __syncthreads();
#pragma unroll
    for (int i = ly; i < 32; i += 8)
        out[(size_t)(c0 + i) * R + r0 + lx] = (__bf16)tile[lx][i];
}

// ---------------- router: wave32 per token ----------------
__global__ __launch_bounds__(256)
void moe_router(const float* __restrict__ x, const float* __restrict__ rwgt,
                int* __restrict__ sel, float* __restrict__ rw,
                int* counts, float* psum, float* zacc) {
    __shared__ float rlds[Ee][Dm];                 // 32 KB
    int tid = threadIdx.x;
    for (int i = tid; i < Ee * Dm; i += 256) rlds[i / Dm][i % Dm] = rwgt[i];
    __syncthreads();

    int wave = tid >> 5, lane = tid & 31;
    int t = blockIdx.x * 8 + wave;

    float acc[Ee];
#pragma unroll
    for (int e = 0; e < Ee; e++) acc[e] = 0.f;
    const float* xr = x + (size_t)t * Dm;
    for (int j = lane; j < Dm; j += 32) {
        float xv = xr[j];
#pragma unroll
        for (int e = 0; e < Ee; e++) acc[e] += xv * rlds[e][j];
    }
#pragma unroll
    for (int e = 0; e < Ee; e++)
        for (int off = 16; off > 0; off >>= 1)
            acc[e] += __shfl_xor(acc[e], off, 32);

    if (lane == 0) {
        float m = acc[0];
#pragma unroll
        for (int e = 1; e < Ee; e++) m = fmaxf(m, acc[e]);
        float se = 0.f, p[Ee];
#pragma unroll
        for (int e = 0; e < Ee; e++) { p[e] = __expf(acc[e] - m); se += p[e]; }
        float inv = 1.f / se;
#pragma unroll
        for (int e = 0; e < Ee; e++) { p[e] *= inv; atomicAdd(psum + e, p[e]); }
        float z = m + logf(se);
        atomicAdd(zacc, z * z);
        // top-2 (ties -> smaller index, matches top_k)
        int e0 = 0; float p0 = p[0];
#pragma unroll
        for (int e = 1; e < Ee; e++) if (p[e] > p0) { p0 = p[e]; e0 = e; }
        int e1 = -1; float p1 = -1.f;
#pragma unroll
        for (int e = 0; e < Ee; e++) if (e != e0 && p[e] > p1) { p1 = p[e]; e1 = e; }
        float s = p0 + p1;
        sel[2 * t]     = e0; rw[2 * t]     = p0 / s;
        sel[2 * t + 1] = e1; rw[2 * t + 1] = p1 / s;
        atomicAdd(counts + e0, 1);
        atomicAdd(counts + e1, 1);
    }
}

// ---------------- scan + loss finalize (tiny, 1 block) ----------------
__global__ void moe_scan_finalize(const int* counts, const float* psum, const float* zacc,
                                  int* off_pad, int* meta, int* blk_exp,
                                  float* out_tail) {
    if (threadIdx.x != 0) return;
    int cum = 0;
    for (int e = 0; e < Ee; e++) {
        off_pad[e] = cum * BLK;
        int bpe = (counts[e] + BLK - 1) / BLK;
        for (int b = 0; b < bpe; b++) blk_exp[cum + b] = e;
        cum += bpe;
    }
    off_pad[Ee] = cum * BLK;
    meta[0] = cum;                  // n_blocks
    meta[1] = cum * BLK;            // cap rows
    float zl = zacc[0] / (float)T;
    float lb = 0.f;
    for (int e = 0; e < Ee; e++)
        lb += (counts[e] / (float)NR) * (psum[e] / (float)T);
    lb *= (float)Ee;
    out_tail[0] = zl;
    out_tail[1] = lb;
    for (int e = 0; e < Ee; e++) out_tail[2 + e] = counts[e] / (float)NR;
}

// ---------------- gather tokens into expert-sorted padded bf16 layout ----------------
__global__ __launch_bounds__(256)
void moe_gather(const float* __restrict__ x, const int* __restrict__ sel,
                const int* __restrict__ off_pad, int* fill,
                int* __restrict__ posmap, __bf16* __restrict__ xg) {
    int wave = threadIdx.x >> 5, lane = threadIdx.x & 31;
    int n = blockIdx.x * 8 + wave;
    int t = n >> 1;
    int e = sel[n];
    int pos = 0;
    if (lane == 0) {
        pos = off_pad[e] + atomicAdd(fill + e, 1);
        posmap[n] = pos;
    }
    pos = __shfl(pos, 0, 32);
    const float* xr = x + (size_t)t * Dm;
    __bf16* dst = xg + (size_t)pos * Dm;
    for (int j = lane; j < Dm; j += 32) dst[j] = (__bf16)xr[j];
}

// ---------------- grouped GEMM: 128x128 tile, 8 waves, WMMA bf16 ----------------
// A: [CAP][ldA] bf16 row-major. B: pre-transposed weights, fragment-ready:
// element (k, colBase+c) of the math B-matrix lives at Bbase[(colBase+c)*ldB + k].
// Both LDS tiles are [idx][k] with contiguous K, double-buffered, filled with
// async global->LDS b128 copies (ASYNCcnt) and consumed by v_wmma bf16.
template <bool GELU>
__global__ __launch_bounds__(256)
void moe_grouped_gemm(const __bf16* __restrict__ A,
                      const __bf16* __restrict__ Bw,
                      __bf16* __restrict__ C,
                      const int* __restrict__ blk_exp,
                      const int* __restrict__ meta,
                      int Kdim, int ldA, int ldB, int ldC,
                      size_t expert_stride) {
    int blk = blockIdx.y;
    if (blk >= meta[0]) return;
    int e = blk_exp[blk];
    const __bf16* Bbase = Bw + (size_t)e * expert_stride;
    int colBase = blockIdx.x * 128;
    int rowBase = blk * 128;

    __shared__ alignas(16) __bf16 Alds[2][128][32];   // [buf][row][k]  16 KB
    __shared__ alignas(16) __bf16 Blds[2][128][32];   // [buf][col][k]  16 KB

    int tid  = threadIdx.x;
    int wave = tid >> 5, lane = tid & 31;
    int fr = lane & 15;      // row (A) / col (B/C) within 16-tile
    int kh = lane >> 4;      // K-half select per ISA lane layout

    // per-thread staging coords: 2x b128 for A, 2x b128 for B per K-step
    int sr = tid >> 1;              // 0..127 (row for A / col for B)
    int sh = (tid & 1) * 16;        // 0 or 16 (k offset)

    v8f acc[8] = {};

    int nsteps = Kdim / 32;
    // prologue: stage K-step 0 into buffer 0
    {
        const __bf16* ga = A + (size_t)(rowBase + sr) * ldA + sh;
        cp_b128(&Alds[0][sr][sh], ga);
        cp_b128(&Alds[0][sr][sh + 8], ga + 8);
        const __bf16* gb = Bbase + (size_t)(colBase + sr) * ldB + sh;
        cp_b128(&Blds[0][sr][sh], gb);
        cp_b128(&Blds[0][sr][sh + 8], gb + 8);
    }

    for (int i = 0; i < nsteps; i++) {
        int cur = i & 1;
        wait_async_all();        // our 4 copies for buffer `cur` have landed
        __syncthreads();         // everyone's copies landed; prev compute done
        if (i + 1 < nsteps) {    // stage next K-step into the other buffer
            int k0 = (i + 1) * 32;
            const __bf16* ga = A + (size_t)(rowBase + sr) * ldA + k0 + sh;
            cp_b128(&Alds[cur ^ 1][sr][sh], ga);
            cp_b128(&Alds[cur ^ 1][sr][sh + 8], ga + 8);
            const __bf16* gb = Bbase + (size_t)(colBase + sr) * ldB + k0 + sh;
            cp_b128(&Blds[cur ^ 1][sr][sh], gb);
            cp_b128(&Blds[cur ^ 1][sr][sh + 8], gb + 8);
        }

        // ---- B fragment: contiguous 16 bf16 per lane (k = 16*kh + s, same order as A) ----
        v16bf bfrag;
        {
            const v4i* bp = (const v4i*)(&Blds[cur][wave * 16 + fr][kh * 16]);
            v4i* d = (v4i*)&bfrag;
            d[0] = bp[0];
            d[1] = bp[1];
        }
#pragma unroll
        for (int rt = 0; rt < 8; rt++) {
            // A fragment: contiguous 16 bf16 at K = kh*16.. (matches B)
            v16bf afrag;
            v4i* d = (v4i*)&afrag;
            d[0] = *(const v4i*)(&Alds[cur][rt * 16 + fr][kh * 16]);
            d[1] = *(const v4i*)(&Alds[cur][rt * 16 + fr][kh * 16 + 8]);
            acc[rt] = __builtin_amdgcn_wmma_f32_16x16x32_bf16(
                false, afrag, false, bfrag, (short)0, acc[rt], false, false);
        }
    }

    // ---- epilogue: (optional exact-erf GELU) + bf16 store ----
#pragma unroll
    for (int rt = 0; rt < 8; rt++) {
#pragma unroll
        for (int i = 0; i < 8; i++) {
            int m = rt * 16 + i + 8 * kh;      // C/D VGPR layout: M = i + 8*(lane/16)
            float v = acc[rt][i];
            if (GELU) v = 0.5f * v * (1.0f + erff(v * 0.70710678118f));
            C[(size_t)(rowBase + m) * ldC + colBase + wave * 16 + fr] = (__bf16)v;
        }
    }
}

// ---------------- combine top-2 (deterministic, no atomics) ----------------
__global__ __launch_bounds__(256)
void moe_combine(const __bf16* __restrict__ Y, const int* __restrict__ posmap,
                 const float* __restrict__ rw, float* __restrict__ out) {
    int t = blockIdx.x;
    int p0 = posmap[2 * t], p1 = posmap[2 * t + 1];
    float w0 = rw[2 * t], w1 = rw[2 * t + 1];
    const __bf16* y0 = Y + (size_t)p0 * Dm;
    const __bf16* y1 = Y + (size_t)p1 * Dm;
    float* o = out + (size_t)t * Dm;
    for (int d = threadIdx.x; d < Dm; d += 256)
        o[d] = w0 * (float)y0[d] + w1 * (float)y1[d];
}

// ---------------- host launcher ----------------
extern "C" void kernel_launch(void* const* d_in, const int* in_sizes, int n_in,
                              void* d_out, int out_size, void* d_ws, size_t ws_size,
                              hipStream_t stream) {
    const float* x    = (const float*)d_in[0];
    const float* w1   = (const float*)d_in[1];
    const float* w2   = (const float*)d_in[2];
    const float* rwgt = (const float*)d_in[3];
    float* out = (float*)d_out;

    char* ws = (char*)d_ws;
    int*    counts  = (int*)  (ws + OFF_COUNTS);
    int*    fill    = (int*)  (ws + OFF_FILL);
    float*  psum    = (float*)(ws + OFF_PSUM);
    float*  zacc    = (float*)(ws + OFF_ZACC);
    int*    off_pad = (int*)  (ws + OFF_OFFPAD);
    int*    meta    = (int*)  (ws + OFF_META);
    int*    blk_exp = (int*)  (ws + OFF_BLKEXP);
    int*    sel     = (int*)  (ws + OFF_SEL);
    float*  rw      = (float*)(ws + OFF_RW);
    int*    posmap  = (int*)  (ws + OFF_POS);
    __bf16* xg      = (__bf16*)(ws + OFF_XG);
    __bf16* w1t     = (__bf16*)(ws + OFF_W1T);
    __bf16* w2t     = (__bf16*)(ws + OFF_W2T);
    __bf16* H       = (__bf16*)(ws + OFF_H);
    __bf16* Y       = (__bf16*)(ws + OFF_Y);

    // 1. init counters, zero padded gathered-X (pad rows feed zeros into WMMA)
    moe_init_small<<<1, 64, 0, stream>>>(counts, fill, psum, zacc);
    {
        long n16 = (long)CAP * Dm * 2 / 16;
        moe_zero_u4<<<(unsigned)((n16 + 255) / 256), 256, 0, stream>>>((uint4*)xg, n16);
    }
    // 2. weights -> bf16, transposed into fragment-ready layout
    //    w1t[(e*DFF+f)*Dm + d] = w1[d][e*DFF+f];  w2t[d*(E*DFF) + e*DFF+f] = w2[e*DFF+f][d]
    moe_transpose_cvt<<<dim3((Ee * DFF) / 32, Dm / 32), 256, 0, stream>>>(w1, w1t, Dm, Ee * DFF);
    moe_transpose_cvt<<<dim3(Dm / 32, (Ee * DFF) / 32), 256, 0, stream>>>(w2, w2t, Ee * DFF, Dm);
    // 3. route
    moe_router<<<T / 8, 256, 0, stream>>>(x, rwgt, sel, rw, counts, psum, zacc);
    // 4. scan offsets + block->expert map + losses into d_out tail
    moe_scan_finalize<<<1, 32, 0, stream>>>(counts, psum, zacc, off_pad, meta, blk_exp,
                                            out + (size_t)T * Dm);
    // 5. gather tokens (f32 -> bf16) into padded expert-sorted layout
    moe_gather<<<NR / 8, 256, 0, stream>>>(x, sel, off_pad, fill, posmap, xg);
    // 6. grouped GEMM1 + GELU: H[cap,DFF] = gelu(Xg @ W1_e)
    moe_grouped_gemm<true><<<dim3(DFF / 128, MAXB), 256, 0, stream>>>(
        xg, w1t, H, blk_exp, meta,
        /*Kdim=*/Dm, /*ldA=*/Dm, /*ldB=*/Dm, /*ldC=*/DFF,
        /*expert_stride=*/(size_t)DFF * Dm);
    // 7. grouped GEMM2: Y[cap,Dm] = H @ W2_e
    moe_grouped_gemm<false><<<dim3(Dm / 128, MAXB), 256, 0, stream>>>(
        H, w2t, Y, blk_exp, meta,
        /*Kdim=*/DFF, /*ldA=*/DFF, /*ldB=*/Ee * DFF, /*ldC=*/Dm,
        /*expert_stride=*/(size_t)DFF);
    // 8. weighted top-2 combine -> out
    moe_combine<<<T, 256, 0, stream>>>(Y, posmap, rw, out);
}
